// LivingMemoryField_85392539779859
// MI455X (gfx1250) — matrix operand order repaired and checked by reading, not compiled, not run on hardware
//
#include <hip/hip_runtime.h>

// ---------- types ----------
typedef __bf16 bf16_t;
typedef __attribute__((ext_vector_type(16))) __bf16 v16bf;
typedef __attribute__((ext_vector_type(8)))  float  v8f;

#define FDIM 4096
#define NEG_INF_F (-1e30f)

// pattern bank geometry (fixed by the reference)
#define P_CONS 10000
#define P_WORK 2000
#define P_TRANS 500
#define ROWPAD_CONS 10000   // multiple of 16
#define ROWPAD_WORK 2000
#define ROWPAD_TRANS 512
#define CHPAD_CONS 10016    // multiple of 32
#define CHPAD_WORK 2016
#define CHPAD_TRANS 512
#define TILES_CONS 625
#define TILES_WORK 125
#define TILES_TRANS 32
#define NCH_CONS 313        // CHPAD/32
#define NCH_WORK 63
#define NCH_TRANS 16

// pull-phase slicing (deterministic partial buffers)
#define SL_CONS 16          // 16 slices x 20 chunks >= 313
#define SL_WORK 4           // 4 slices x 16 chunks >= 63
#define SL_TRANS 2          // 2 slices x 8 chunks = 16
#define NSLICE (SL_CONS + SL_WORK + SL_TRANS)   // 22

__device__ __forceinline__ bf16_t f2bf(float f) {
    union { float f; unsigned u; } in; in.f = f;
    unsigned r = in.u + 0x7FFFu + ((in.u >> 16) & 1u);   // RNE
    union { unsigned short s; bf16_t b; } out; out.s = (unsigned short)(r >> 16);
    return out.b;
}

// permuted position inside a 32-chunk for the WMMA A-operand staging:
// layout [k0..15 | k16..31] (contiguous)
__device__ __forceinline__ int permpos(int j) {
    return j;
}

// ---------- one-time preamble kernels ----------

// gates: sigmoid(tags @ reg) * mask  (trans gate = mask)
__global__ void gates_kernel(const float* __restrict__ ctags, const float* __restrict__ wtags,
                             const float* __restrict__ reg,
                             const unsigned char* __restrict__ cmask,
                             const unsigned char* __restrict__ wmask,
                             const unsigned char* __restrict__ tmask,
                             float* gc, float* gw, float* gt) {
    __shared__ float r[64];
    if (threadIdx.x < 64) r[threadIdx.x] = reg[threadIdx.x];
    __syncthreads();
    int i = blockIdx.x * blockDim.x + threadIdx.x;
    if (i < P_CONS) {
        float a = 0.f;
        #pragma unroll 8
        for (int j = 0; j < 64; ++j) a += ctags[(size_t)i * 64 + j] * r[j];
        gc[i] = cmask[i] ? (1.f / (1.f + __expf(-a))) : 0.f;
    } else if (i < P_CONS + P_WORK) {
        int p = i - P_CONS;
        float a = 0.f;
        #pragma unroll 8
        for (int j = 0; j < 64; ++j) a += wtags[(size_t)p * 64 + j] * r[j];
        gw[p] = wmask[p] ? (1.f / (1.f + __expf(-a))) : 0.f;
    } else if (i < P_CONS + P_WORK + P_TRANS) {
        int p = i - P_CONS - P_WORK;
        gt[p] = tmask[p] ? 1.f : 0.f;
    }
}

// row-major fp32 -> bf16 (zero pad rows)
__global__ void conv_bf16_kernel(const float* __restrict__ src, bf16_t* __restrict__ dst,
                                 int P, int RowPad) {
    size_t total = (size_t)RowPad * FDIM;
    for (size_t i = (size_t)blockIdx.x * blockDim.x + threadIdx.x; i < total;
         i += (size_t)gridDim.x * blockDim.x) {
        size_t row = i >> 12;
        dst[i] = (row < (size_t)P) ? f2bf(src[i]) : f2bf(0.f);
    }
}

// tiled transpose fp32[P,4096] -> bf16[4096,Ppad] (zero pad columns)
__global__ void transpose_bf16_kernel(const float* __restrict__ src, bf16_t* __restrict__ dst,
                                      int P, int Ppad) {
    __shared__ float tile[32][33];
    int p0 = blockIdx.x * 32, k0 = blockIdx.y * 32;
    int tx = threadIdx.x, ty = threadIdx.y;
    #pragma unroll
    for (int i = 0; i < 4; ++i) {
        int p = p0 + ty + 8 * i;
        tile[ty + 8 * i][tx] = (p < P) ? src[(size_t)p * FDIM + k0 + tx] : 0.f;
    }
    __syncthreads();
    #pragma unroll
    for (int i = 0; i < 4; ++i) {
        int k = k0 + ty + 8 * i;
        dst[(size_t)k * Ppad + p0 + tx] = f2bf(tile[tx][ty + 8 * i]);
    }
}

// ---------- per-step kernels ----------

// t[r] = sum_i s[i] * V[i,r]   (16 blocks, one per rank)
__global__ void rank_kernel(const float* __restrict__ s, const float* __restrict__ V,
                            float* __restrict__ t) {
    __shared__ float red[256];
    int r = blockIdx.x;
    float a = 0.f;
    for (int i = threadIdx.x; i < FDIM; i += 256) a += s[i] * V[(size_t)i * 16 + r];
    red[threadIdx.x] = a;
    __syncthreads();
    for (int o = 128; o > 0; o >>= 1) {
        if (threadIdx.x < o) red[threadIdx.x] += red[threadIdx.x + o];
        __syncthreads();
    }
    if (threadIdx.x == 0) t[r] = red[0];
}

// s_ev = clip(A_diag)*s + U@t + ext(step0); also stage permuted bf16 broadcast copy
__global__ void sev_kernel(const float* __restrict__ s, const float* __restrict__ Adiag,
                           const float* __restrict__ U, const float* __restrict__ t,
                           const float* __restrict__ ext, int step,
                           float* __restrict__ s_ev, bf16_t* __restrict__ sbc) {
    __shared__ float ts[16];
    if (threadIdx.x < 16) ts[threadIdx.x] = t[threadIdx.x];
    __syncthreads();
    int i = blockIdx.x * blockDim.x + threadIdx.x;
    float a = fminf(fmaxf(Adiag[i], 0.01f), 0.999f);
    float acc = a * s[i];
    #pragma unroll
    for (int r = 0; r < 16; ++r) acc += ts[r] * U[(size_t)i * 16 + r];
    if (step == 0) acc += ext[i];
    s_ev[i] = acc;
    int c = i >> 5, j = i & 31;
    sbc[c * 32 + permpos(j)] = f2bf(acc);
}

// scores via WMMA: one wave per (16-row tile, K-half). A = broadcast(s_ev), B = P rows.
// D[0, n] (lane n, acc[0]) = partial dot(P[row n], s_ev). Partials written to
// disjoint half-buffers -> deterministic (no atomics).
__global__ __launch_bounds__(32) void scores_kernel(
        const bf16_t* __restrict__ Pb0, const bf16_t* __restrict__ Pb1,
        const bf16_t* __restrict__ Pb2,
        float* __restrict__ sc0, float* __restrict__ sc1, float* __restrict__ sc2,
        const bf16_t* __restrict__ sbc) {
    int tb = blockIdx.x;
    const bf16_t* Pb; float* sc; int P, Ppad; float beta;
    if (tb < TILES_CONS) {
        Pb = Pb0; sc = sc0; P = P_CONS;  Ppad = CHPAD_CONS;  beta = 1.f;
    } else if (tb < TILES_CONS + TILES_WORK) {
        tb -= TILES_CONS;
        Pb = Pb1; sc = sc1; P = P_WORK;  Ppad = CHPAD_WORK;  beta = 2.f;
    } else {
        tb -= TILES_CONS + TILES_WORK;
        Pb = Pb2; sc = sc2; P = P_TRANS; Ppad = CHPAD_TRANS; beta = 4.f;
    }
    int kh = blockIdx.y;                     // K-half: 0 or 1 (64 chunks each)
    int lane = threadIdx.x, lg = lane >> 4, ll = lane & 15;
    const bf16_t* prow = Pb + (size_t)(tb * 16 + ll) * FDIM + lg * 16;
    const bf16_t* arow = sbc + lg * 16;
    v8f c = {};
    int c0 = kh * 64, c1 = c0 + 64;
    #pragma unroll 4
    for (int ch = c0; ch < c1; ++ch) {
        v16bf a = *(const v16bf*)(arow + (size_t)ch * 32);
        v16bf b = *(const v16bf*)(prow + (size_t)ch * 32);
        c = __builtin_amdgcn_wmma_f32_16x16x32_bf16(false, a, false, b, (short)0, c,
                                                    false, false);
    }
    int row = tb * 16 + ll;
    if (lane < 16 && row < P) sc[(size_t)kh * Ppad + row] = beta * c[0];
}

// masked softmax over (half0 + half1) scores
// -> coef[p] = w_l * attn[p] * gate[p] * depth[p], packed permuted bf16
__global__ __launch_bounds__(1024) void softmax_kernel(
        const float* __restrict__ lw,
        const float* __restrict__ sc0, const float* __restrict__ sc1,
        const float* __restrict__ sc2,
        const float* __restrict__ g0, const float* __restrict__ g1,
        const float* __restrict__ g2,
        const float* __restrict__ d0, const float* __restrict__ d1,
        const float* __restrict__ d2,
        bf16_t* __restrict__ cf0, bf16_t* __restrict__ cf1, bf16_t* __restrict__ cf2) {
    int b = blockIdx.x;
    const float* sc; const float* g; const float* dep; bf16_t* cf; int P, Ppad;
    if (b == 0)      { sc = sc0; g = g0; dep = d0; cf = cf0; P = P_CONS;  Ppad = CHPAD_CONS; }
    else if (b == 1) { sc = sc1; g = g1; dep = d1; cf = cf1; P = P_WORK;  Ppad = CHPAD_WORK; }
    else             { sc = sc2; g = g2; dep = d2; cf = cf2; P = P_TRANS; Ppad = CHPAD_TRANS; }
    int tid = threadIdx.x;
    __shared__ float red[1024];
    float m = -3.4e38f;
    for (int p = tid; p < P; p += 1024) {
        float v = (g[p] > 0.f) ? (sc[p] + sc[(size_t)Ppad + p]) : NEG_INF_F;
        m = fmaxf(m, v);
    }
    red[tid] = m; __syncthreads();
    for (int o = 512; o > 0; o >>= 1) {
        if (tid < o) red[tid] = fmaxf(red[tid], red[tid + o]);
        __syncthreads();
    }
    float smax = red[0]; __syncthreads();
    float su = 0.f;
    for (int p = tid; p < P; p += 1024) {
        float v = (g[p] > 0.f) ? (sc[p] + sc[(size_t)Ppad + p]) : NEG_INF_F;
        su += __expf(v - smax);
    }
    red[tid] = su; __syncthreads();
    for (int o = 512; o > 0; o >>= 1) {
        if (tid < o) red[tid] += red[tid + o];
        __syncthreads();
    }
    float ssum = red[0];
    // layer-weight softmax (3 values)
    float l0 = lw[0], l1 = lw[1], l2 = lw[2];
    float lm = fmaxf(l0, fmaxf(l1, l2));
    float e0 = __expf(l0 - lm), e1 = __expf(l1 - lm), e2 = __expf(l2 - lm);
    float wl = ((b == 0) ? e0 : (b == 1) ? e1 : e2) / (e0 + e1 + e2);
    float inv = wl / ssum;
    for (int p = tid; p < Ppad; p += 1024) {
        float v = 0.f;
        if (p < P && g[p] > 0.f) {
            float s = sc[p] + sc[(size_t)Ppad + p];
            v = __expf(s - smax) * inv * g[p] * dep[p];
        }
        int c = p >> 5, j = p & 31;
        cf[c * 32 + permpos(j)] = f2bf(v);
    }
}

// pull via WMMA: A = broadcast(coef chunk), B = P^T tile -> D[0,n] = partial pull.
// Each (column-tile, slice) wave writes a private slice row -> deterministic.
__global__ __launch_bounds__(32) void pull_kernel(
        const bf16_t* __restrict__ PT0, const bf16_t* __restrict__ PT1,
        const bf16_t* __restrict__ PT2,
        const bf16_t* __restrict__ cf0, const bf16_t* __restrict__ cf1,
        const bf16_t* __restrict__ cf2,
        float* __restrict__ pull_part) {
    int ct = blockIdx.x;   // column tile: 0..255
    int y  = blockIdx.y;   // slice: 0..15 cons, 16..19 work, 20..21 trans
    const bf16_t* PT; const bf16_t* cf; int Ppad, c0, c1;
    if (y < SL_CONS) {
        PT = PT0; cf = cf0; Ppad = CHPAD_CONS;
        c0 = y * 20; c1 = (c0 + 20 < NCH_CONS) ? c0 + 20 : NCH_CONS;
    } else if (y < SL_CONS + SL_WORK) {
        PT = PT1; cf = cf1; Ppad = CHPAD_WORK;
        int s = y - SL_CONS;
        c0 = s * 16; c1 = (c0 + 16 < NCH_WORK) ? c0 + 16 : NCH_WORK;
    } else {
        PT = PT2; cf = cf2; Ppad = CHPAD_TRANS;
        int s = y - SL_CONS - SL_WORK;
        c0 = s * 8; c1 = c0 + 8;
    }
    int lane = threadIdx.x, lg = lane >> 4, ll = lane & 15;
    const bf16_t* prow = PT + (size_t)(ct * 16 + ll) * Ppad + lg * 16;
    const bf16_t* arow = cf + lg * 16;
    v8f c = {};
    #pragma unroll 4
    for (int ch = c0; ch < c1; ++ch) {
        v16bf a = *(const v16bf*)(arow + (size_t)ch * 32);
        v16bf b = *(const v16bf*)(prow + (size_t)ch * 32);
        c = __builtin_amdgcn_wmma_f32_16x16x32_bf16(false, a, false, b, (short)0, c,
                                                    false, false);
    }
    if (lane < 16) pull_part[(size_t)y * FDIM + ct * 16 + ll] = c[0];
}

// s_new = (1-lambda)*s_ev + sum_slices(pull_part) + sigma*noise[step]
__global__ void combine_kernel(const float* __restrict__ s_ev,
                               const float* __restrict__ pull_part,
                               const float* __restrict__ noise, int step,
                               float* __restrict__ out) {
    int i = blockIdx.x * blockDim.x + threadIdx.x;
    float p = 0.f;
    #pragma unroll
    for (int y = 0; y < NSLICE; ++y) p += pull_part[(size_t)y * FDIM + i];
    out[i] = 0.99f * s_ev[i] + p + 0.01f * noise[(size_t)step * FDIM + i];
}

// ---------- host launcher ----------
extern "C" void kernel_launch(void* const* d_in, const int* in_sizes, int n_in,
                              void* d_out, int out_size, void* d_ws, size_t ws_size,
                              hipStream_t stream) {
    (void)in_sizes; (void)n_in; (void)out_size; (void)ws_size;

    const float* field_state = (const float*)d_in[0];
    const float* ext         = (const float*)d_in[1];
    const float* A_diag      = (const float*)d_in[2];
    const float* U           = (const float*)d_in[3];
    const float* V           = (const float*)d_in[4];
    const float* lw          = (const float*)d_in[5];
    const float* consP       = (const float*)d_in[6];
    const float* consD       = (const float*)d_in[7];
    const float* consT       = (const float*)d_in[8];
    const float* workP       = (const float*)d_in[9];
    const float* workD       = (const float*)d_in[10];
    const float* workT       = (const float*)d_in[11];
    const float* transP      = (const float*)d_in[12];
    const float* transD      = (const float*)d_in[13];
    const float* reg         = (const float*)d_in[14];
    const float* noise       = (const float*)d_in[15];
    const unsigned char* cmask = (const unsigned char*)d_in[16];
    const unsigned char* wmask = (const unsigned char*)d_in[17];
    const unsigned char* tmask = (const unsigned char*)d_in[18];
    float* out = (float*)d_out;

    // workspace carve-up (256B aligned)
    char* base = (char*)d_ws;
    size_t off = 0;
    auto carve = [&](size_t bytes) -> char* {
        char* p = base + off;
        off += (bytes + 255) & ~(size_t)255;
        return p;
    };
    float*  s_ws = (float*)carve(FDIM * 4);
    float*  s_ev = (float*)carve(FDIM * 4);
    float*  ppart = (float*)carve((size_t)NSLICE * FDIM * 4);
    float*  t_ws = (float*)carve(16 * 4);
    bf16_t* sbc  = (bf16_t*)carve(FDIM * 2);
    float*  g0   = (float*)carve(CHPAD_CONS * 4);
    float*  g1   = (float*)carve(CHPAD_WORK * 4);
    float*  g2   = (float*)carve(CHPAD_TRANS * 4);
    float*  sc0  = (float*)carve(2u * CHPAD_CONS * 4);   // two K-halves
    float*  sc1  = (float*)carve(2u * CHPAD_WORK * 4);
    float*  sc2  = (float*)carve(2u * CHPAD_TRANS * 4);
    bf16_t* cf0  = (bf16_t*)carve(CHPAD_CONS * 2);
    bf16_t* cf1  = (bf16_t*)carve(CHPAD_WORK * 2);
    bf16_t* cf2  = (bf16_t*)carve(CHPAD_TRANS * 2);
    bf16_t* Pb0  = (bf16_t*)carve((size_t)ROWPAD_CONS * FDIM * 2);
    bf16_t* Pb1  = (bf16_t*)carve((size_t)ROWPAD_WORK * FDIM * 2);
    bf16_t* Pb2  = (bf16_t*)carve((size_t)ROWPAD_TRANS * FDIM * 2);
    bf16_t* PT0  = (bf16_t*)carve((size_t)FDIM * CHPAD_CONS * 2);
    bf16_t* PT1  = (bf16_t*)carve((size_t)FDIM * CHPAD_WORK * 2);
    bf16_t* PT2  = (bf16_t*)carve((size_t)FDIM * CHPAD_TRANS * 2);

    // ---- preamble: gates + bf16 staging (runs every call; graph-capture safe) ----
    gates_kernel<<<(P_CONS + P_WORK + P_TRANS + 255) / 256, 256, 0, stream>>>(
        consT, workT, reg, cmask, wmask, tmask, g0, g1, g2);
    conv_bf16_kernel<<<4096, 256, 0, stream>>>(consP,  Pb0, P_CONS,  ROWPAD_CONS);
    conv_bf16_kernel<<<2048, 256, 0, stream>>>(workP,  Pb1, P_WORK,  ROWPAD_WORK);
    conv_bf16_kernel<<<1024, 256, 0, stream>>>(transP, Pb2, P_TRANS, ROWPAD_TRANS);
    transpose_bf16_kernel<<<dim3(NCH_CONS,  FDIM / 32), dim3(32, 8), 0, stream>>>(
        consP,  PT0, P_CONS,  CHPAD_CONS);
    transpose_bf16_kernel<<<dim3(NCH_WORK,  FDIM / 32), dim3(32, 8), 0, stream>>>(
        workP,  PT1, P_WORK,  CHPAD_WORK);
    transpose_bf16_kernel<<<dim3(NCH_TRANS, FDIM / 32), dim3(32, 8), 0, stream>>>(
        transP, PT2, P_TRANS, CHPAD_TRANS);

    // ---- 8 sequential steps ----
    for (int step = 0; step < 8; ++step) {
        const float* sPrev = (step == 0) ? field_state : s_ws;
        rank_kernel<<<16, 256, 0, stream>>>(sPrev, V, t_ws);
        sev_kernel<<<16, 256, 0, stream>>>(sPrev, A_diag, U, t_ws, ext, step,
                                           s_ev, sbc);
        scores_kernel<<<dim3(TILES_CONS + TILES_WORK + TILES_TRANS, 2), 32, 0, stream>>>(
            Pb0, Pb1, Pb2, sc0, sc1, sc2, sbc);
        softmax_kernel<<<3, 1024, 0, stream>>>(lw, sc0, sc1, sc2, g0, g1, g2,
                                               consD, workD, transD, cf0, cf1, cf2);
        pull_kernel<<<dim3(FDIM / 16, NSLICE), 32, 0, stream>>>(PT0, PT1, PT2,
                                                                cf0, cf1, cf2, ppart);
        combine_kernel<<<16, 256, 0, stream>>>(s_ev, ppart, noise, step,
                                               (step == 7) ? out : s_ws);
    }
}
